// OIntInferMatMul_90443421319762
// MI455X (gfx1250) — 8-bit matrix operand order repaired and checked by reading, compiled, not run on hardware
//
#include <hip/hip_runtime.h>


typedef int v8i __attribute__((ext_vector_type(8)));
typedef int v4i __attribute__((ext_vector_type(4)));

#define TILE_M 128
#define TILE_N 128
#define KDIM   64
#define SEQ    2048
#define NBH    32
#define FRAG_BYTES 1024
#define QA_BYTES   ((size_t)NBH * SEQ * KDIM)
#define QB_BYTES   ((size_t)NBH * SEQ * KDIM)
#define ROW_DW 17

__global__ __launch_bounds__(256)
void quant_swizzle_kernel(const float* __restrict__ x1,
                          const float* __restrict__ x2,
                          const float* __restrict__ s1ll,
                          const float* __restrict__ sx1,
                          const float* __restrict__ s2ll,
                          const float* __restrict__ sx2,
                          unsigned int* __restrict__ qA,
                          unsigned int* __restrict__ qB)
{
    const int d = blockIdx.x * 256 + threadIdx.x;
    const int bh = d >> 15;
    const int r  = d & 32767;

    if (blockIdx.y == 0) {
        const float rat = s1ll[0] / sx1[0];
        const int m  = r >> 4;
        const int kd = r & 15;
        const float4 v = *(const float4*)(x1 + (size_t)bh * SEQ * KDIM
                                             + (size_t)m * KDIM + kd * 4);
        const int q0 = __float2int_rn(__builtin_amdgcn_fmed3f(v.x * rat, -128.0f, 127.0f));
        const int q1 = __float2int_rn(__builtin_amdgcn_fmed3f(v.y * rat, -128.0f, 127.0f));
        const int q2 = __float2int_rn(__builtin_amdgcn_fmed3f(v.z * rat, -128.0f, 127.0f));
        const int q3 = __float2int_rn(__builtin_amdgcn_fmed3f(v.w * rat, -128.0f, 127.0f));
        const unsigned pk = (unsigned)(q0 & 255) | ((unsigned)(q1 & 255) << 8)
                          | ((unsigned)(q2 & 255) << 16) | ((unsigned)(q3 & 255) << 24);
        const int mBlock = m >> 4, m16 = m & 15;
        const int group = kd >> 2, rem = kd & 3;
        const int half = rem >> 1, pair = rem & 1;
        const int i = group * 2 + pair;
        const int lane = half * 16 + m16;
        qA[((size_t)(bh * 128 + mBlock) * FRAG_BYTES >> 2) + lane * 8 + i] = pk;
    } else {
        const float rat = s2ll[0] / sx2[0];
        const int n  = r & 2047;
        const int kd = r >> 11;
        const int k  = kd * 4;
        const float* B = x2 + (size_t)bh * KDIM * SEQ;
        const float b0 = B[(size_t)(k + 0) * SEQ + n];
        const float b1 = B[(size_t)(k + 1) * SEQ + n];
        const float b2 = B[(size_t)(k + 2) * SEQ + n];
        const float b3 = B[(size_t)(k + 3) * SEQ + n];
        const int q0 = __float2int_rn(__builtin_amdgcn_fmed3f(b0 * rat, -128.0f, 127.0f));
        const int q1 = __float2int_rn(__builtin_amdgcn_fmed3f(b1 * rat, -128.0f, 127.0f));
        const int q2 = __float2int_rn(__builtin_amdgcn_fmed3f(b2 * rat, -128.0f, 127.0f));
        const int q3 = __float2int_rn(__builtin_amdgcn_fmed3f(b3 * rat, -128.0f, 127.0f));
        const unsigned pk = (unsigned)(q0 & 255) | ((unsigned)(q1 & 255) << 8)
                          | ((unsigned)(q2 & 255) << 16) | ((unsigned)(q3 & 255) << 24);
        const int nBlock = n >> 4, n16 = n & 15;
        const int group2 = kd >> 2, rem = kd & 3;
        const int halfK = rem >> 1, within = rem & 1;
        const int i = group2 * 2 + within;
        const int lane = halfK * 16 + n16;
        qB[((size_t)(bh * 128 + nBlock) * FRAG_BYTES >> 2) + lane * 8 + i] = pk;
    }
}

__global__ __launch_bounds__(256)
void wmma_kernel(const unsigned int* __restrict__ qA,
                 const unsigned int* __restrict__ qB,
                 float* __restrict__ out)
{
    const int tid  = threadIdx.x;
    const int nT   = blockIdx.x;
    const int mT   = blockIdx.y;
    const int bh   = blockIdx.z;
    const int wave = tid >> 5;
    const int lane = tid & 31;
    const int half = lane >> 4;
    const int l16  = lane & 15;

    const int mBlock = mT * 8 + wave;
    const unsigned int* ap = qA + ((size_t)(bh * 128 + mBlock) * FRAG_BYTES >> 2) + lane * 8;
    const v4i a_lo = *(const v4i*)(ap);
    const v4i a_hi = *(const v4i*)(ap + 4);
    const v8i afrag = __builtin_shufflevector(a_lo, a_hi, 0, 1, 2, 3, 4, 5, 6, 7);

    float* C = out + (size_t)bh * SEQ * SEQ
                   + (size_t)(mT * TILE_M + wave * 16) * SEQ + (size_t)nT * TILE_N;

    #pragma unroll
    for (int nb = 0; nb < 8; ++nb) {
        const int nBlock = nT * 8 + nb;
        const unsigned int* bp = qB + ((size_t)(bh * 128 + nBlock) * FRAG_BYTES >> 2) + lane * 8;
        const v4i b_lo = *(const v4i*)(bp);
        const v4i b_hi = *(const v4i*)(bp + 4);
        const v8i bfrag = __builtin_shufflevector(b_lo, b_hi, 0, 1, 2, 3, 4, 5, 6, 7);

        v8i acc = {};
        acc = __builtin_amdgcn_wmma_i32_16x16x64_iu8(true, afrag, true, bfrag,
                                                     acc, false, false);

        #pragma unroll
        for (int r = 0; r < 8; ++r) {
            const int m = r + half * 8;
            const int n = nb * 16 + l16;
            const float y = __builtin_amdgcn_fmed3f(
                rintf((float)acc[r] * 0.0625f), -128.0f, 127.0f);
            C[(size_t)m * SEQ + n] = y;
        }
    }
}

__global__ __launch_bounds__(256)
void fused_fallback_kernel(const float* __restrict__ x1,
                           const float* __restrict__ x2,
                           const float* __restrict__ s1ll,
                           const float* __restrict__ sx1,
                           const float* __restrict__ s2ll,
                           const float* __restrict__ sx2,
                           float* __restrict__ out)
{
    __shared__ unsigned int ldsA [TILE_M * ROW_DW];
    __shared__ unsigned int ldsBT[TILE_N * ROW_DW];

    const int tid = threadIdx.x;
    const int nT = blockIdx.x, mT = blockIdx.y, bh = blockIdx.z;
    const float r1 = s1ll[0] / sx1[0];
    const float r2 = s2ll[0] / sx2[0];

    const float* A = x1 + (size_t)bh * SEQ * KDIM + (size_t)mT * TILE_M * KDIM;
    const float* B = x2 + (size_t)bh * KDIM * SEQ + (size_t)nT * TILE_N;
    float*       C = out + (size_t)bh * SEQ * SEQ
                         + (size_t)mT * TILE_M * SEQ + (size_t)nT * TILE_N;

    #pragma unroll
    for (int it = 0; it < 8; ++it) {
        const int d = tid + it * 256, m = d >> 4, kd = d & 15;
        const float4 v = *(const float4*)(A + m * KDIM + kd * 4);
        const int q0 = __float2int_rn(__builtin_amdgcn_fmed3f(v.x * r1, -128.0f, 127.0f));
        const int q1 = __float2int_rn(__builtin_amdgcn_fmed3f(v.y * r1, -128.0f, 127.0f));
        const int q2 = __float2int_rn(__builtin_amdgcn_fmed3f(v.z * r1, -128.0f, 127.0f));
        const int q3 = __float2int_rn(__builtin_amdgcn_fmed3f(v.w * r1, -128.0f, 127.0f));
        ldsA[m * ROW_DW + kd] = (unsigned)(q0 & 255) | ((unsigned)(q1 & 255) << 8)
                              | ((unsigned)(q2 & 255) << 16) | ((unsigned)(q3 & 255) << 24);
    }
    #pragma unroll
    for (int it = 0; it < 8; ++it) {
        const int d = tid + it * 256, n = d & 127, kd = d >> 7, k = kd * 4;
        const float b0 = B[(size_t)(k + 0) * SEQ + n];
        const float b1 = B[(size_t)(k + 1) * SEQ + n];
        const float b2 = B[(size_t)(k + 2) * SEQ + n];
        const float b3 = B[(size_t)(k + 3) * SEQ + n];
        const int q0 = __float2int_rn(__builtin_amdgcn_fmed3f(b0 * r2, -128.0f, 127.0f));
        const int q1 = __float2int_rn(__builtin_amdgcn_fmed3f(b1 * r2, -128.0f, 127.0f));
        const int q2 = __float2int_rn(__builtin_amdgcn_fmed3f(b2 * r2, -128.0f, 127.0f));
        const int q3 = __float2int_rn(__builtin_amdgcn_fmed3f(b3 * r2, -128.0f, 127.0f));
        ldsBT[n * ROW_DW + kd] = (unsigned)(q0 & 255) | ((unsigned)(q1 & 255) << 8)
                               | ((unsigned)(q2 & 255) << 16) | ((unsigned)(q3 & 255) << 24);
    }
    __syncthreads();

    const int wave = tid >> 5, lane = tid & 31, half = lane >> 4, l16 = lane & 15;
    const int mBase = wave * 16;

    v8i afrag;
    #pragma unroll
    for (int i = 0; i < 8; ++i) {
        const int kd = (i >> 1) * 4 + half * 2 + (i & 1);
        afrag[i] = (int)ldsA[(mBase + l16) * ROW_DW + kd];
    }
    #pragma unroll
    for (int nb = 0; nb < 8; ++nb) {
        v8i bfrag;
        #pragma unroll
        for (int i = 0; i < 8; ++i) {
            const int kd = (i >> 1) * 4 + half * 2 + (i & 1);
            bfrag[i] = (int)ldsBT[(nb * 16 + l16) * ROW_DW + kd];
        }
        v8i acc = {};
        acc = __builtin_amdgcn_wmma_i32_16x16x64_iu8(true, afrag, true, bfrag,
                                                     acc, false, false);
        #pragma unroll
        for (int r = 0; r < 8; ++r) {
            const float y = __builtin_amdgcn_fmed3f(
                rintf((float)acc[r] * 0.0625f), -128.0f, 127.0f);
            C[(size_t)(mBase + r + half * 8) * SEQ + (nb * 16 + l16)] = y;
        }
    }
}

extern "C" void kernel_launch(void* const* d_in, const int* in_sizes, int n_in,
                              void* d_out, int out_size, void* d_ws, size_t ws_size,
                              hipStream_t stream) {
    (void)in_sizes; (void)n_in; (void)out_size;
    const float* x1   = (const float*)d_in[0];
    const float* x2   = (const float*)d_in[1];
    const float* s1ll = (const float*)d_in[2];
    const float* sx1  = (const float*)d_in[3];
    const float* s2ll = (const float*)d_in[4];
    const float* sx2  = (const float*)d_in[5];
    float* out = (float*)d_out;

    if (ws_size >= QA_BYTES + QB_BYTES && d_ws != nullptr) {
        unsigned int* qA = (unsigned int*)d_ws;
        unsigned int* qB = (unsigned int*)((char*)d_ws + QA_BYTES);
        quant_swizzle_kernel<<<dim3(4096, 2, 1), dim3(256), 0, stream>>>(
            x1, x2, s1ll, sx1, s2ll, sx2, qA, qB);
        wmma_kernel<<<dim3(SEQ / TILE_N, SEQ / TILE_M, NBH), dim3(256), 0, stream>>>(
            qA, qB, out);
    } else {
        fused_fallback_kernel<<<dim3(SEQ / TILE_N, SEQ / TILE_M, NBH),
                                dim3(256), 0, stream>>>(
            x1, x2, s1ll, sx1, s2ll, sx2, out);
    }
}
